// GNNEnocder_13271448945097
// MI455X (gfx1250) — matrix operand order repaired and checked by reading, not compiled, not run on hardware
//
#include <hip/hip_runtime.h>
#include <hip/hip_bf16.h>

// ---------------------------------------------------------------------------
// GCN (3x GCNConv + PairNorm + ReLU) for MI455X / gfx1250.
// GEMMs: bf16 WMMA (v_wmma_f32_16x16x32_bf16, f32 accum), 32 WMMAs per wave,
// W pre-packed per layer into the wave32 B-fragment layout (bf16, 32 KB).
// Edge aggregation: wave-per-edge float4 gather + f32 atomic scatter
// (h and accumulator are L2-resident: 2 x 51 MB < 192 MB L2).
// ---------------------------------------------------------------------------

#define GCN_N 100000
#define GCN_E 3200000
#define GCN_D 128
#define GCN_EPS 1e-8f

typedef __attribute__((ext_vector_type(16))) __bf16 v16bf;
typedef __attribute__((ext_vector_type(8)))  float  v8f;

// ---------------------------------------------------------------------------
// Degree / dinv
// ---------------------------------------------------------------------------
__global__ void k_init_deg(float* __restrict__ deg, int n) {
    int i = blockIdx.x * blockDim.x + threadIdx.x;
    if (i < n) deg[i] = 1.0f;  // self-loop contributes 1 to every node's degree
}

__global__ void k_accum_deg(const int* __restrict__ dst, float* __restrict__ deg, int e) {
    int stride = gridDim.x * blockDim.x;
    for (int i = blockIdx.x * blockDim.x + threadIdx.x; i < e; i += stride)
        atomicAdd(&deg[dst[i]], 1.0f);
}

__global__ void k_finish_dinv(float* __restrict__ deg, int n) {
    int i = blockIdx.x * blockDim.x + threadIdx.x;
    if (i < n) {
        float d = deg[i];
        deg[i] = (d > 0.0f) ? rsqrtf(d) : 0.0f;
    }
}

// ---------------------------------------------------------------------------
// Pack W[128x128] f32 (row-major, W[k][n]) into bf16 B-fragments:
//   Wp[((s*8 + t)*32 + lane)*16 + kp]  with
//   n = t*16 + (lane&15), k = s*32 + (lane>>4)*8 + (kp&7) + (kp>>3)*16
// so each lane's v16bf B fragment is one coalesced 32-byte load.
// (CDNA5 ISA 7.12.2: 16-bit B 32x16 — lanes 0-15 K=0..7,16..23, lanes 16-31 K=8..15,24..31.)
// ---------------------------------------------------------------------------
__global__ void k_convW(const float* __restrict__ W, __bf16* __restrict__ Wp) {
    int idx = blockIdx.x * blockDim.x + threadIdx.x;   // 0 .. 16383
    if (idx < 4 * 8 * 32 * 16) {
        int kp   = idx & 15;
        int lane = (idx >> 4) & 31;
        int t    = (idx >> 9) & 7;
        int s    = idx >> 12;
        int n    = t * 16 + (lane & 15);
        int k    = s * 32 + (lane >> 4) * 8 + (kp & 7) + (kp >> 3) * 16;
        Wp[idx]  = (__bf16)W[k * GCN_D + n];
    }
}

// ---------------------------------------------------------------------------
// GEMM: H[n x 128] = X[n x 128] @ W, one wave per 16-row strip.
// Per K-step (32): build A fragment once, issue 8 WMMAs (one per column tile).
// 32 v_wmma per wave; A reuse x8; B loads are packed v16bf.
// ---------------------------------------------------------------------------
__global__ void __launch_bounds__(256)
k_gemm_wmma(const float* __restrict__ X, const __bf16* __restrict__ Wp,
            float* __restrict__ H, int nrows) {
    const int wave    = threadIdx.x >> 5;                       // 0..7
    const int lane    = threadIdx.x & 31;
    const int half    = lane >> 4;
    const int l16     = lane & 15;
    const int rowTile = blockIdx.x * 8 + wave;
    if (rowTile * 16 >= nrows) return;

    const int row = rowTile * 16 + l16;                         // A row for this lane
    const v16bf* __restrict__ Bfrag = (const v16bf*)Wp;         // [(s*8+t)*32 + lane]

    v8f acc[8] = {};

    #pragma unroll
    for (int s = 0; s < 4; ++s) {
        const int k0 = s * 32;
        v16bf a;
        #pragma unroll
        for (int v = 0; v < 8; ++v) {
            // A 16x32 bf16 layout: lanes 0-15 hold K {0..7,16..23}, lanes 16-31 K {8..15,24..31}
            int ka = k0 + (v < 4 ? 2 * v : 8 + 2 * v) + half * 8;
            float2 xa = *(const float2*)(X + row * GCN_D + ka);
            a[2 * v]     = (__bf16)xa.x;
            a[2 * v + 1] = (__bf16)xa.y;
        }
        #pragma unroll
        for (int t = 0; t < 8; ++t) {
            v16bf b = Bfrag[(s * 8 + t) * 32 + lane];
            acc[t] = __builtin_amdgcn_wmma_f32_16x16x32_bf16(
                false, a, false, b, (short)0, acc[t], false, false);
        }
    }

    // C/D 32-bit 16x16 layout: VGPR v -> row (half*8 + v), column l16 of tile t.
    #pragma unroll
    for (int t = 0; t < 8; ++t) {
        #pragma unroll
        for (int v = 0; v < 8; ++v) {
            int cM = rowTile * 16 + half * 8 + v;
            H[cM * GCN_D + t * 16 + l16] = acc[t][v];
        }
    }
}

// ---------------------------------------------------------------------------
// Aggregation init: out = bias + dinv[i]^2 * h[i]   (self-loop message)
// ---------------------------------------------------------------------------
__global__ void k_init_agg(const float* __restrict__ h, const float* __restrict__ dinv,
                           const float* __restrict__ bias, float* __restrict__ out, int n) {
    int i = blockIdx.x * blockDim.x + threadIdx.x;
    if (i < n * GCN_D) {
        int r = i >> 7;
        int c = i & (GCN_D - 1);
        float dv = dinv[r];
        out[i] = bias[c] + dv * dv * h[i];
    }
}

// ---------------------------------------------------------------------------
// Edge scatter: one wave per edge. Each lane moves 4 features (float4 gather,
// 4x global_atomic_add_f32). h and out are L2-resident.
// ---------------------------------------------------------------------------
__global__ void k_edge_scatter(const int* __restrict__ src, const int* __restrict__ dst,
                               const float* __restrict__ dinv, const float* __restrict__ h,
                               float* __restrict__ out, int e) {
    const int lane   = threadIdx.x & 31;
    const int wid    = (blockIdx.x * blockDim.x + threadIdx.x) >> 5;
    const int nwaves = (gridDim.x * blockDim.x) >> 5;

    for (int ed = wid; ed < e; ed += nwaves) {
        int s = src[ed];
        int d = dst[ed];
        float nm = dinv[s] * dinv[d];
        float4 hv = *(const float4*)(h + s * GCN_D + lane * 4);
        float* o = out + d * GCN_D + lane * 4;
        atomicAdd(o + 0, hv.x * nm);
        atomicAdd(o + 1, hv.y * nm);
        atomicAdd(o + 2, hv.z * nm);
        atomicAdd(o + 3, hv.w * nm);
    }
}

// ---------------------------------------------------------------------------
// PairNorm 'PN': x -= colmean; x /= (sqrt(mean row ||x||^2) + eps); fused ReLU.
// stats[0..127]: colsum -> (after finalize) colmean; stats[128]: sum(x^2);
// stats[129]: 1/scale.
// ---------------------------------------------------------------------------
__global__ void k_zero_stats(float* __restrict__ stats) {
    if (threadIdx.x < 132) stats[threadIdx.x] = 0.0f;
}

__global__ void k_pn_reduce(const float* __restrict__ x, float* __restrict__ stats, int n) {
    __shared__ float ssq[GCN_D];
    int c = threadIdx.x;  // 128 threads, one per column
    float colsum = 0.0f, sq = 0.0f;
    for (int r = blockIdx.x; r < n; r += gridDim.x) {
        float v = x[r * GCN_D + c];
        colsum += v;
        sq += v * v;
    }
    atomicAdd(&stats[c], colsum);
    ssq[c] = sq;
    __syncthreads();
    for (int s = 64; s > 0; s >>= 1) {
        if (c < s) ssq[c] += ssq[c + s];
        __syncthreads();
    }
    if (c == 0) atomicAdd(&stats[GCN_D], ssq[0]);
}

__global__ void k_pn_finalize(float* __restrict__ stats, int n) {
    __shared__ float sm[GCN_D];
    int j = threadIdx.x;  // 128 threads
    float m = stats[j] / (float)n;
    sm[j] = m * m;
    __syncthreads();
    for (int s = 64; s > 0; s >>= 1) {
        if (j < s) sm[j] += sm[j + s];
        __syncthreads();
    }
    stats[j] = m;  // overwrite colsum with mean
    if (j == 0) {
        // sum_ij (x-m)^2 = sum(x^2) - N * sum_j m_j^2
        float tot = stats[GCN_D] - (float)n * sm[0];
        float rn  = sqrtf(fmaxf(tot / (float)n, 0.0f)) + GCN_EPS;
        stats[GCN_D + 1] = 1.0f / rn;
    }
}

__global__ void k_pn_apply_relu(const float* __restrict__ x, const float* __restrict__ stats,
                                float* __restrict__ y, int n) {
    int i = blockIdx.x * blockDim.x + threadIdx.x;
    if (i < n * GCN_D) {
        int c = i & (GCN_D - 1);
        float v = (x[i] - stats[c]) * stats[GCN_D + 1];
        y[i] = fmaxf(v, 0.0f);
    }
}

// ---------------------------------------------------------------------------
// Launch
// ---------------------------------------------------------------------------
extern "C" void kernel_launch(void* const* d_in, const int* in_sizes, int n_in,
                              void* d_out, int out_size, void* d_ws, size_t ws_size,
                              hipStream_t stream) {
    (void)in_sizes; (void)n_in; (void)out_size; (void)ws_size;

    const float* x   = (const float*)d_in[0];
    const int*   ei  = (const int*)d_in[1];   // [2, E]: row 0 = src, row 1 = dst
    const float* W1  = (const float*)d_in[2];
    const float* b1  = (const float*)d_in[3];
    const float* W2  = (const float*)d_in[4];
    const float* b2  = (const float*)d_in[5];
    const float* W3  = (const float*)d_in[6];
    const float* b3  = (const float*)d_in[7];
    float* out = (float*)d_out;

    const int N = GCN_N, E = GCN_E;
    const int* src = ei;
    const int* dst = ei + E;

    // Workspace carve-up: dinv[N] f32 | stats[256] f32 | Wp[16384] bf16 | bufA | bufB
    float*  dinv  = (float*)d_ws;
    float*  stats = dinv + N;
    __bf16* Wp    = (__bf16*)(stats + 256);
    float*  bufA  = (float*)(Wp + 16384);
    float*  bufB  = bufA + (size_t)N * GCN_D;

    const dim3 blk256(256);
    const dim3 grdN((N + 255) / 256);
    const dim3 grdND((N * GCN_D + 255) / 256);
    const int  nRowTiles = N / 16;                    // N % 16 == 0
    const dim3 gemmGrid((nRowTiles + 7) / 8);         // 8 waves (strips) per block
    const dim3 gemmBlk(256);
    const dim3 convGrid(64);                          // 16384 elements
    const dim3 edgeGrid(4096);                        // 32768 waves, grid-stride
    const dim3 pnGrid(1024);
    const dim3 pnBlk(GCN_D);

    // Degree / dinv (self-loops folded in via init to 1)
    k_init_deg<<<grdN, blk256, 0, stream>>>(dinv, N);
    k_accum_deg<<<edgeGrid, blk256, 0, stream>>>(dst, dinv, E);
    k_finish_dinv<<<grdN, blk256, 0, stream>>>(dinv, N);

    // ---- Layer 1: gemm(x -> bufA); agg(bufA -> bufB); pn+relu(bufB -> bufA)
    k_convW<<<convGrid, blk256, 0, stream>>>(W1, Wp);
    k_gemm_wmma<<<gemmGrid, gemmBlk, 0, stream>>>(x, Wp, bufA, N);
    k_init_agg<<<grdND, blk256, 0, stream>>>(bufA, dinv, b1, bufB, N);
    k_edge_scatter<<<edgeGrid, blk256, 0, stream>>>(src, dst, dinv, bufA, bufB, E);
    k_zero_stats<<<1, 256, 0, stream>>>(stats);
    k_pn_reduce<<<pnGrid, pnBlk, 0, stream>>>(bufB, stats, N);
    k_pn_finalize<<<1, pnBlk, 0, stream>>>(stats, N);
    k_pn_apply_relu<<<grdND, blk256, 0, stream>>>(bufB, stats, bufA, N);

    // ---- Layer 2: gemm(bufA -> bufB); agg(bufB -> bufA); pn+relu(bufA -> bufB)
    k_convW<<<convGrid, blk256, 0, stream>>>(W2, Wp);
    k_gemm_wmma<<<gemmGrid, gemmBlk, 0, stream>>>(bufA, Wp, bufB, N);
    k_init_agg<<<grdND, blk256, 0, stream>>>(bufB, dinv, b2, bufA, N);
    k_edge_scatter<<<edgeGrid, blk256, 0, stream>>>(src, dst, dinv, bufB, bufA, E);
    k_zero_stats<<<1, 256, 0, stream>>>(stats);
    k_pn_reduce<<<pnGrid, pnBlk, 0, stream>>>(bufA, stats, N);
    k_pn_finalize<<<1, pnBlk, 0, stream>>>(stats, N);
    k_pn_apply_relu<<<grdND, blk256, 0, stream>>>(bufA, stats, bufB, N);

    // ---- Layer 3: gemm(bufB -> bufA); agg(bufA -> d_out)  (no PairNorm/ReLU)
    k_convW<<<convGrid, blk256, 0, stream>>>(W3, Wp);
    k_gemm_wmma<<<gemmGrid, gemmBlk, 0, stream>>>(bufB, Wp, bufA, N);
    k_init_agg<<<grdND, blk256, 0, stream>>>(bufA, dinv, b3, out, N);
    k_edge_scatter<<<edgeGrid, blk256, 0, stream>>>(src, dst, dinv, bufA, out, E);
}
